// SimpleRNN_45303315038863
// MI455X (gfx1250) — matrix operand order repaired and checked by reading, not compiled, not run on hardware
//
#include <hip/hip_runtime.h>
#include <math.h>

typedef __attribute__((ext_vector_type(16))) _Float16     v16h;
typedef __attribute__((ext_vector_type(8)))  float        v8f;
typedef __attribute__((ext_vector_type(4)))  unsigned int u32x4;
typedef __attribute__((ext_vector_type(2)))  float        f32x2;

#define B_SZ  4096
#define T_SZ  512
#define DIN   10
#define HSZ   20
#define DOUT  2
#define WAVES_PER_WG 8
#define LROW  32   // padded h-row length in f16 (pad cols 20..31 stay zero)

// gfx1250 has native V_TANH_F32 (confirmed in round-2 asm).
__device__ __forceinline__ float fast_tanh(float x) {
#if __has_builtin(__builtin_amdgcn_tanhf)
    return __builtin_amdgcn_tanhf(x);
#else
    float xc = fminf(fmaxf(x, -9.0f), 9.0f);
    float z  = __builtin_amdgcn_exp2f(xc * 2.8853900817779268f);
    return (z - 1.0f) * __builtin_amdgcn_rcpf(z + 1.0f);
#endif
}

// A-layout (16-bit 16x32): lane l (hid=l>>4, m=l&15):
//   e in 0..7  -> k = e + 8*hid ;  e in 8..15 -> k = e + 8 + 8*hid
// B-layout (16-bit 32x16): lane l, elem e -> (k as in A-layout, n = l&15 [+16 tile1])
// C/D layout (f32 16x16):  lane l, vgpr v -> (m = v + 8*hid, n = l&15 [+16 tile1])

__global__ __launch_bounds__(WAVES_PER_WG * 32)
void rnn2_fused_wmma(const float* __restrict__ x,
                     const float* __restrict__ w_ih0, const float* __restrict__ w_hh0,
                     const float* __restrict__ b_ih0, const float* __restrict__ b_hh0,
                     const float* __restrict__ w_ih1, const float* __restrict__ w_hh1,
                     const float* __restrict__ b_ih1, const float* __restrict__ b_hh1,
                     const float* __restrict__ fc_w,  const float* __restrict__ fc_b,
                     float* __restrict__ out)
{
    __shared__ _Float16 hbuf[WAVES_PER_WG][2][16][LROW];

    const int lane = threadIdx.x & 31;
    const int wv   = threadIdx.x >> 5;
    const int hid  = lane >> 4;
    const int ml   = lane & 15;
    const int b0   = (blockIdx.x * WAVES_PER_WG + wv) * 16;

    _Float16* h0row = &hbuf[wv][0][0][0];
    _Float16* h1row = &hbuf[wv][1][0][0];

    // Zero this wave's LDS slice: initial hidden state AND the k>=20 pad.
    {
        u32x4 z = (u32x4)0u;
        u32x4* zp = (u32x4*)h0row;              // 2*16*32*2B = 2048B = 128 x u32x4
        #pragma unroll
        for (int i = 0; i < 4; ++i) zp[lane + 32 * i] = z;
    }

    // ---- Pre-gather weights into per-lane B-layout registers ----
    v16h B0[2], B1a[2], B1b[2];
    v8f  C0[2], C1[2];
    #pragma unroll
    for (int tl = 0; tl < 2; ++tl) {
        const int n = ml + 16 * tl;
        #pragma unroll
        for (int e = 0; e < 16; ++e) {
            const int k = (e < 8) ? (e + 8 * hid) : (e + 8 + 8 * hid);
            float w0 = 0.f, wa = 0.f, wb = 0.f;
            if (n < HSZ) {
                if (k < HSZ)                 w0 = w_hh0[n * HSZ + k];
                else if (k < HSZ + DIN)      w0 = w_ih0[n * DIN + (k - HSZ)];
                if (k < HSZ) { wa = w_ih1[n * HSZ + k]; wb = w_hh1[n * HSZ + k]; }
            }
            B0[tl][e]  = (_Float16)w0;
            B1a[tl][e] = (_Float16)wa;
            B1b[tl][e] = (_Float16)wb;
        }
        const float bias0 = (n < HSZ) ? (b_ih0[n] + b_hh0[n]) : 0.f;
        const float bias1 = (n < HSZ) ? (b_ih1[n] + b_hh1[n]) : 0.f;
        #pragma unroll
        for (int v = 0; v < 8; ++v) { C0[tl][v] = bias0; C1[tl][v] = bias1; }
    }

    const _Float16* h0rd = h0row + ml * LROW;
    const _Float16* h1rd = h1row + ml * LROW;

    // Branch-free per-lane x slice: hid0 consumes floats 0..3, hid1 floats 4..9.
    // Both halves load 3 x f32x2 starting at (4*hid); always in-bounds (row = 10 floats).
    const float* xptr = x + ((size_t)b0 + ml) * (size_t)T_SZ * DIN + 4 * hid;
    f32x2 xc0 = *(const f32x2*)(xptr + 0);
    f32x2 xc1 = *(const f32x2*)(xptr + 2);
    f32x2 xc2 = *(const f32x2*)(xptr + 4);

    for (int t = 0; t < T_SZ; ++t) {
        asm volatile("" ::: "memory");
        // ---------- Layer 0: A = [h0_prev | x_t], one wmma per N-tile ----------
        union { v16h v; u32x4 q[2]; } a0;
        a0.q[0] = *(const u32x4*)(h0rd + 8 * hid);
        a0.q[1] = *(const u32x4*)(h0rd + 16 + 8 * hid);
        if (hid == 0) {   // k = 20..23 -> e12..15
            a0.v[12] = (_Float16)xc0.x; a0.v[13] = (_Float16)xc0.y;
            a0.v[14] = (_Float16)xc1.x; a0.v[15] = (_Float16)xc1.y;
        } else {          // k = 24..29 -> e8..13
            a0.v[8]  = (_Float16)xc0.x; a0.v[9]  = (_Float16)xc0.y;
            a0.v[10] = (_Float16)xc1.x; a0.v[11] = (_Float16)xc1.y;
            a0.v[12] = (_Float16)xc2.x; a0.v[13] = (_Float16)xc2.y;
        }
        // Software-pipelined prefetch of x_{t+1} (clamped); its wait lands next
        // iteration, hiding the global-load latency behind this whole step.
        {
            const int tn = (t + 1 < T_SZ) ? (t + 1) : t;
            const float* xn = xptr + (size_t)tn * DIN;
            xc0 = *(const f32x2*)(xn + 0);
            xc1 = *(const f32x2*)(xn + 2);
            xc2 = *(const f32x2*)(xn + 4);
        }

        v8f d0 = __builtin_amdgcn_wmma_f32_16x16x32_f16(false, a0.v, false, B0[0], (short)0, C0[0], false, false);
        v8f d1 = __builtin_amdgcn_wmma_f32_16x16x32_f16(false, a0.v, false, B0[1], (short)0, C0[1], false, false);

        #pragma unroll
        for (int v = 0; v < 8; ++v)
            h0row[(v + 8 * hid) * LROW + ml] = (_Float16)fast_tanh(d0[v]);
        if (ml < 4) {
            #pragma unroll
            for (int v = 0; v < 8; ++v)
                h0row[(v + 8 * hid) * LROW + 16 + ml] = (_Float16)fast_tanh(d1[v]);
        }
        asm volatile("" ::: "memory");

        // ---------- Layer 1: two chained wmmas per N-tile ----------
        union { v16h v; u32x4 q[2]; } a1, a2;
        a1.q[0] = *(const u32x4*)(h0rd + 8 * hid);
        a1.q[1] = *(const u32x4*)(h0rd + 16 + 8 * hid);
        a2.q[0] = *(const u32x4*)(h1rd + 8 * hid);
        a2.q[1] = *(const u32x4*)(h1rd + 16 + 8 * hid);

        v8f e0 = __builtin_amdgcn_wmma_f32_16x16x32_f16(false, a1.v, false, B1a[0], (short)0, C1[0], false, false);
        e0     = __builtin_amdgcn_wmma_f32_16x16x32_f16(false, a2.v, false, B1b[0], (short)0, e0,    false, false);
        v8f e1 = __builtin_amdgcn_wmma_f32_16x16x32_f16(false, a1.v, false, B1a[1], (short)0, C1[1], false, false);
        e1     = __builtin_amdgcn_wmma_f32_16x16x32_f16(false, a2.v, false, B1b[1], (short)0, e1,    false, false);

        #pragma unroll
        for (int v = 0; v < 8; ++v)
            h1row[(v + 8 * hid) * LROW + ml] = (_Float16)fast_tanh(e0[v]);
        if (ml < 4) {
            #pragma unroll
            for (int v = 0; v < 8; ++v)
                h1row[(v + 8 * hid) * LROW + 16 + ml] = (_Float16)fast_tanh(e1[v]);
        }
    }
    asm volatile("" ::: "memory");

    // ---------- FC head: lane (hid, ml) -> out[b0+ml][hid], DOUT == 2 ----------
    {
        float acc = fc_b[hid];
        const _Float16* hr = h1row + ml * LROW;
        #pragma unroll
        for (int j = 0; j < HSZ; ++j)
            acc += (float)hr[j] * fc_w[hid * HSZ + j];
        out[((size_t)b0 + ml) * DOUT + hid] = acc;
    }
}

extern "C" void kernel_launch(void* const* d_in, const int* in_sizes, int n_in,
                              void* d_out, int out_size, void* d_ws, size_t ws_size,
                              hipStream_t stream) {
    (void)in_sizes; (void)n_in; (void)out_size; (void)d_ws; (void)ws_size;
    const float* xp    = (const float*)d_in[0];
    const float* wih0  = (const float*)d_in[1];
    const float* whh0  = (const float*)d_in[2];
    const float* bih0  = (const float*)d_in[3];
    const float* bhh0  = (const float*)d_in[4];
    const float* wih1  = (const float*)d_in[5];
    const float* whh1  = (const float*)d_in[6];
    const float* bih1  = (const float*)d_in[7];
    const float* bhh1  = (const float*)d_in[8];
    const float* fcw   = (const float*)d_in[9];
    const float* fcb   = (const float*)d_in[10];
    float* outp        = (float*)d_out;

    dim3 grid(B_SZ / (16 * WAVES_PER_WG));   // 32 blocks
    dim3 block(WAVES_PER_WG * 32);           // 256 threads = 8 waves
    rnn2_fused_wmma<<<grid, block, 0, stream>>>(xp, wih0, whh0, bih0, bhh0,
                                                wih1, whh1, bih1, bhh1,
                                                fcw, fcb, outp);
}
